// GPN_3633542333121
// MI455X (gfx1250) — matrix operand order repaired and checked by reading, not compiled, not run on hardware
//
#include <hip/hip_runtime.h>
#include <cstdint>
#include <cstddef>

#define Hc 128
#define Bc 32
#define Nc 4096
#define Ec 16384

typedef _Float16 half_t;
typedef __attribute__((ext_vector_type(16))) _Float16 v16h;
typedef __attribute__((ext_vector_type(8)))  float    v8f;
typedef unsigned int u32x4 __attribute__((ext_vector_type(4)));
typedef int          i32x4 __attribute__((ext_vector_type(4)));
typedef int          i32x8 __attribute__((ext_vector_type(8)));

#if defined(__has_builtin)
#if __has_builtin(__builtin_amdgcn_tensor_load_to_lds)
#define HAVE_TDM 1
#endif
#endif

__device__ __forceinline__ float sigf(float x) { return 1.0f / (1.0f + __expf(-x)); }

// low 32 bits of a generic shared pointer == LDS byte offset (flat LDS aperture)
__device__ __forceinline__ uint32_t lds_offset_of(const void* p) {
  return (uint32_t)(uintptr_t)p;
}

// ---------------------------------------------------------------------------
// TDM: DMA a rows x cols f16 2-D tile (row-major, stride == cols) into LDS.
// D# layout per CDNA5 ISA sec.8. Tracked by TENSORcnt.
// ---------------------------------------------------------------------------
#ifdef HAVE_TDM
__device__ __forceinline__ void tdm_load_f16_2d(const void* gptr, uint32_t lds_off,
                                                int rows, int cols) {
  uint64_t ga = (uint64_t)(uintptr_t)gptr;
  u32x4 g0;
  g0[0] = 1u;                                          // count=1, user descriptor
  g0[1] = lds_off;                                     // lds_addr (bytes)
  g0[2] = (uint32_t)(ga & 0xFFFFFFFFu);                // global_addr[31:0]
  g0[3] = (uint32_t)((ga >> 32) & 0x01FFFFFFu)         // global_addr[56:32]
          | (2u << 30);                                // type=2 ("image")
  i32x8 g1;
  g1[0] = (1 << 16);                                   // wg_mask=0, data_size=1 (2 bytes)
  g1[1] = (cols & 0xFFFF) << 16;                       // tensor_dim0[15:0]
  g1[2] = ((cols >> 16) & 0xFFFF) | ((rows & 0xFFFF) << 16);   // dim0[31:16] | dim1[15:0]
  g1[3] = ((rows >> 16) & 0xFFFF) | ((cols & 0xFFFF) << 16);   // dim1[31:16] | tile_dim0
  g1[4] = (rows & 0xFFFF);                             // tile_dim1 (tile_dim2 = 0)
  g1[5] = cols;                                        // tensor_dim0_stride[31:0]
  g1[6] = 0;                                           // stride0 hi | stride1 lo (unused, 2-D)
  g1[7] = 0;
  i32x4 z4 = { 0, 0, 0, 0 };
#if __clang_major__ >= 23
  i32x8 z8 = { 0, 0, 0, 0, 0, 0, 0, 0 };
  __builtin_amdgcn_tensor_load_to_lds(g0, g1, z4, z4, z8, 0);
#else
  __builtin_amdgcn_tensor_load_to_lds(g0, g1, z4, z4, 0);
#endif
}
#endif

// stage one 128x128 f16 matrix into LDS (TDM if available, else cooperative copy)
__device__ __forceinline__ void stage_weight(half_t* dst, const half_t* src) {
#ifdef HAVE_TDM
  if (threadIdx.x < 32) {                 // wave 0 only issues the DMA
    tdm_load_f16_2d(src, lds_offset_of(dst), Hc, Hc);
  }
#else
  for (int i = threadIdx.x; i < (Hc * Hc) / 8; i += blockDim.x)
    ((uint4*)dst)[i] = ((const uint4*)src)[i];
#endif
}

__device__ __forceinline__ void stage_wait() {
#ifdef HAVE_TDM
  if (threadIdx.x < 32) __builtin_amdgcn_s_wait_tensorcnt(0);
#endif
  __syncthreads();
}

// ---------------------------------------------------------------------------
// Small elementwise / reduction kernels
// ---------------------------------------------------------------------------
__global__ void f32_to_f16_kernel(const float* __restrict__ in, half_t* __restrict__ out, int n) {
  int i = blockIdx.x * blockDim.x + threadIdx.x;
  if (i < n) out[i] = (half_t)in[i];
}

__global__ void zero_f32_kernel(float* __restrict__ p, size_t n) {
  size_t stride = (size_t)gridDim.x * blockDim.x;
  for (size_t i = (size_t)blockIdx.x * blockDim.x + threadIdx.x; i < n; i += stride) p[i] = 0.0f;
}

__global__ void embed_nodes_kernel(const float* __restrict__ X,
                                   const float* __restrict__ W, const float* __restrict__ bias,
                                   float* __restrict__ out32, half_t* __restrict__ out16) {
  size_t row = blockIdx.x;           // b*N + n
  int h = threadIdx.x;               // 128 threads
  float x0 = X[row * 2 + 0], x1 = X[row * 2 + 1];
  float v = fmaf(W[h * 2 + 0], x0, fmaf(W[h * 2 + 1], x1, bias[h]));
  out32[row * Hc + h] = v;
  out16[row * Hc + h] = (half_t)v;
}

__global__ void embed_edges_kernel(const float* __restrict__ EF,
                                   const float* __restrict__ W, const float* __restrict__ bias,
                                   float* __restrict__ FE) {
  size_t row = blockIdx.x;           // b*E + e
  int h = threadIdx.x;
  const float* f = EF + row * 4;
  float v = bias[h];
  #pragma unroll
  for (int k = 0; k < 4; ++k) v = fmaf(W[h * 4 + k], f[k], v);
  FE[row * Hc + h] = v;
}

__global__ void node_mean_kernel(const float* __restrict__ node32, float* __restrict__ meanx) {
  int b = blockIdx.x, h = threadIdx.x;   // 128 threads
  float s = 0.f;
  for (int n = 0; n < Nc; ++n) s += node32[((size_t)b * Nc + n) * Hc + h];
  meanx[b * Hc + h] = s * (1.0f / Nc);
}

__global__ void scatter_nb_kernel(const float* __restrict__ node32, const int* __restrict__ EI,
                                  float* __restrict__ nb32, float* __restrict__ cnt) {
  size_t be = blockIdx.x;            // b*E + e
  int b = (int)(be / Ec), e = (int)(be % Ec);
  int u = EI[(size_t)b * 2 * Ec + e];
  int v = EI[(size_t)b * 2 * Ec + Ec + e];
  int h = threadIdx.x;               // 128 threads
  size_t ru = ((size_t)b * Nc + u) * Hc;
  size_t rv = ((size_t)b * Nc + v) * Hc;
  float uval = node32[ru + h];
  float vval = node32[rv + h];
  atomicAdd(&nb32[ru + h], vval);
  atomicAdd(&nb32[rv + h], uval);
  if (h == 0) {
    atomicAdd(&cnt[(size_t)b * Nc + u], 1.0f);
    atomicAdd(&cnt[(size_t)b * Nc + v], 1.0f);
  }
}

__global__ void normalize_nb_kernel(const float* __restrict__ nb32, const float* __restrict__ cnt,
                                    half_t* __restrict__ nb16) {
  size_t row = blockIdx.x;
  int h = threadIdx.x;
  float c = fmaxf(cnt[row], 1.0f);
  nb16[row * Hc + h] = (half_t)(nb32[row * Hc + h] / c);
}

// ---------------------------------------------------------------------------
// Fused dual-GEMM ctx layer: out = r*(A @ W^T + bW) + (1-r)*relu(nb @ G^T + bG)
// Weights DMA'd to LDS via TENSOR_LOAD_TO_LDS; B fragments via ds_load_b128.
// ---------------------------------------------------------------------------
__global__ __launch_bounds__(256) void ctx_layer_kernel(
    const half_t* __restrict__ A16, const half_t* __restrict__ N16,
    const half_t* __restrict__ W16, const half_t* __restrict__ G16,
    const float* __restrict__ bW, const float* __restrict__ bG,
    const float* __restrict__ rptr,
    float* __restrict__ out32, half_t* __restrict__ out16) {
  __shared__ half_t ldsW[Hc * Hc];    // 32 KB
  __shared__ half_t ldsG[Hc * Hc];    // 32 KB
  stage_weight(ldsW, W16);
  stage_weight(ldsG, G16);

  const int wave = threadIdx.x >> 5;
  const int lane = threadIdx.x & 31;
  const int lrow = lane & 15;
  const int lhalf = (lane < 16) ? 0 : 1;
  const int rowBase = blockIdx.x * 128 + wave * 16;
  const float rr = rptr[0];

  v8f acc1[8] = {};
  v8f acc2[8] = {};

  const half_t* rowA = A16 + (size_t)(rowBase + lrow) * Hc;
  const half_t* rowN = N16 + (size_t)(rowBase + lrow) * Hc;
  __builtin_prefetch(rowA + (size_t)128 * Hc, 0, 0);   // -> global_prefetch_b8
  __builtin_prefetch(rowN + (size_t)128 * Hc, 0, 0);

  stage_wait();                        // s_wait_tensorcnt 0 + barrier

  #pragma unroll
  for (int ks = 0; ks < 4; ++ks) {
    union { v16h v; uint4 q[2]; } a1, a2;
    const int h0 = lhalf * 8;                       // A frag: k = j+{0|8}, j+{16|24}
    a1.q[0] = *(const uint4*)(rowA + ks * 32 + h0);
    a1.q[1] = *(const uint4*)(rowA + ks * 32 + h0 + 16);
    a2.q[0] = *(const uint4*)(rowN + ks * 32 + h0);
    a2.q[1] = *(const uint4*)(rowN + ks * 32 + h0 + 16);
    #pragma unroll
    for (int nt = 0; nt < 8; ++nt) {
      union { v16h v; uint4 q[2]; } bw, bg;
      const int n = nt * 16 + lrow;                 // B frag: n = lane%16, k = j+{0|8}, j+{16|24}
      const half_t* pw = ldsW + (size_t)n * Hc + ks * 32 + lhalf * 8;
      const half_t* pg = ldsG + (size_t)n * Hc + ks * 32 + lhalf * 8;
      bw.q[0] = *(const uint4*)(pw);  bw.q[1] = *(const uint4*)(pw + 16);   // ds_load_b128
      bg.q[0] = *(const uint4*)(pg);  bg.q[1] = *(const uint4*)(pg + 16);
      acc1[nt] = __builtin_amdgcn_wmma_f32_16x16x32_f16(false, a1.v, false, bw.v,
                                                        (short)0, acc1[nt], false, false);
      acc2[nt] = __builtin_amdgcn_wmma_f32_16x16x32_f16(false, a2.v, false, bg.v,
                                                        (short)0, acc2[nt], false, false);
    }
  }

  // C/D layout: VGPR r holds M = r + 8*lhalf, N = lane%16
  #pragma unroll
  for (int nt = 0; nt < 8; ++nt) {
    const int n = nt * 16 + lrow;
    const float c1 = bW[n], c2 = bG[n];
    #pragma unroll
    for (int r = 0; r < 8; ++r) {
      const int m = rowBase + r + lhalf * 8;
      float v1 = acc1[nt][r] + c1;
      float v2 = acc2[nt][r] + c2;
      float val = rr * v1 + (1.0f - rr) * fmaxf(v2, 0.0f);
      out32[(size_t)m * Hc + n] = val;
      out16[(size_t)m * Hc + n] = (half_t)val;
    }
  }
}

__global__ void gather_final_edge_kernel(const int* __restrict__ EI, const float* __restrict__ ctx,
                                         float* __restrict__ FE) {
  size_t be = blockIdx.x;
  int b = (int)(be / Ec), e = (int)(be % Ec);
  int u = EI[(size_t)b * 2 * Ec + e];
  int v = EI[(size_t)b * 2 * Ec + Ec + e];
  int h = threadIdx.x;
  float add = ctx[((size_t)b * Nc + u) * Hc + h] + ctx[((size_t)b * Nc + v) * Hc + h];
  FE[be * Hc + h] += add;
}

__global__ void mask_sum_kernel(const int* __restrict__ mask, int* __restrict__ firstb) {
  __shared__ int sd[256];
  int b = blockIdx.x, tid = threadIdx.x;
  int s = 0;
  for (int n = tid; n < Nc; n += 256) s += mask[(size_t)b * Nc + n];
  sd[tid] = s; __syncthreads();
  for (int k = 128; k > 0; k >>= 1) { if (tid < k) sd[tid] += sd[tid + k]; __syncthreads(); }
  if (tid == 0) firstb[b] = (sd[0] == 0) ? 1 : 0;
}

// ---------------------------------------------------------------------------
// LSTM (2 steps) + q projection. One block of 128 threads per batch element.
// ---------------------------------------------------------------------------
struct LstmP {
  const float *Wxi, *bxi, *Whi, *bhi, *wci, *bci;
  const float *Wxf, *bxf, *Whf, *bhf, *wcf, *bcf;
  const float *Wxc, *bxc, *Whc, *bhc;
  const float *Wxo, *bxo, *Who, *bho, *wco, *bco;
  const float *Wq, *h0, *c0;
};

__device__ __forceinline__ float dot128(const float* __restrict__ W, const float* __restrict__ v, int row) {
  float s = 0.f;
  #pragma unroll 8
  for (int k = 0; k < Hc; ++k) s = fmaf(W[row * Hc + k], v[k], s);
  return s;
}

__device__ void lstm_step_dev(const LstmP& P, const float* xs, float* hs, float* cs, int t) {
  float i = sigf(dot128(P.Wxi, xs, t) + P.bxi[t] + dot128(P.Whi, hs, t) + P.bhi[t] +
                 dot128(P.wci, cs, t) + P.bci[t]);
  float f = sigf(dot128(P.Wxf, xs, t) + P.bxf[t] + dot128(P.Whf, hs, t) + P.bhf[t] +
                 dot128(P.wcf, cs, t) + P.bcf[t]);
  float g = tanhf(dot128(P.Wxc, xs, t) + P.bxc[t] + dot128(P.Whc, hs, t) + P.bhc[t]);
  float cn = f * cs[t] + i * g;
  float oxh = dot128(P.Wxo, xs, t) + P.bxo[t] + dot128(P.Who, hs, t) + P.bho[t];
  __syncthreads();
  cs[t] = cn;                   // o-gate reads the *updated* c vector
  __syncthreads();
  float o = sigf(oxh + dot128(P.wco, cs, t) + P.bco[t]);
  float hn = o * tanhf(cn);
  __syncthreads();
  hs[t] = hn;
  __syncthreads();
}

__global__ __launch_bounds__(128) void lstm_kernel(LstmP P, const float* __restrict__ meanx,
                                                   const float* __restrict__ FE,
                                                   const int* __restrict__ lastIdx,
                                                   float* __restrict__ hOut, float* __restrict__ cOut,
                                                   float* __restrict__ qproj) {
  __shared__ float xs[Hc], hs[Hc], cs[Hc];
  int b = blockIdx.x, t = threadIdx.x;
  xs[t] = meanx[b * Hc + t];
  hs[t] = P.h0[t];
  cs[t] = P.c0[t];
  __syncthreads();
  lstm_step_dev(P, xs, hs, cs, t);
  int li = lastIdx[b];
  __syncthreads();
  xs[t] = FE[((size_t)b * Ec + li) * Hc + t];
  __syncthreads();
  lstm_step_dev(P, xs, hs, cs, t);
  hOut[b * Hc + t] = hs[t];
  cOut[b * Hc + t] = cs[t];
  float q = 0.f;
  #pragma unroll 8
  for (int k = 0; k < Hc; ++k) q = fmaf(P.Wq[t * Hc + k], hs[k], q);
  qproj[b * Hc + t] = q;
}

// ---------------------------------------------------------------------------
// Fused attention: u[b,e] = sum_n v[n] * tanh(q[b,n] + (FE @ Wref^T)[b,e,n])
// Wref DMA'd to LDS (TDM); FE f32 -> f16 fragments in registers; no ref_proj in HBM.
// ---------------------------------------------------------------------------
__global__ __launch_bounds__(256) void attn_kernel(const float* __restrict__ FE,
                                                   const half_t* __restrict__ Wr16,
                                                   const float* __restrict__ qproj,
                                                   const float* __restrict__ vvec,
                                                   float* __restrict__ ulog) {
  __shared__ half_t ldsW[Hc * Hc];    // 32 KB
  stage_weight(ldsW, Wr16);

  const int wave = threadIdx.x >> 5;
  const int lane = threadIdx.x & 31;
  const int lrow = lane & 15;
  const int lhalf = (lane < 16) ? 0 : 1;
  const int rowBase = blockIdx.x * 128 + wave * 16;     // flat edge row b*E + e
  const int b = rowBase / Ec;                           // 128 | E -> tile stays in one b

  v8f acc[8] = {};
  const float* rowF = FE + (size_t)(rowBase + lrow) * Hc;
  __builtin_prefetch(rowF + (size_t)128 * Hc, 0, 0);

  stage_wait();

  #pragma unroll
  for (int ks = 0; ks < 4; ++ks) {
    union { v16h v; half_t h[16]; } a;
    const float* p = rowF + ks * 32;
    const int h0 = lhalf * 8;
    #pragma unroll
    for (int j = 0; j < 8; ++j) a.h[j] = (half_t)p[h0 + j];
    #pragma unroll
    for (int j = 0; j < 8; ++j) a.h[8 + j] = (half_t)p[h0 + 16 + j];
    #pragma unroll
    for (int nt = 0; nt < 8; ++nt) {
      union { v16h v; uint4 q[2]; } bw;
      const int n = nt * 16 + lrow;
      const half_t* pw = ldsW + (size_t)n * Hc + ks * 32 + lhalf * 8;
      bw.q[0] = *(const uint4*)(pw);  bw.q[1] = *(const uint4*)(pw + 16);
      acc[nt] = __builtin_amdgcn_wmma_f32_16x16x32_f16(false, a.v, false, bw.v,
                                                       (short)0, acc[nt], false, false);
    }
  }

  float sum[8];
  #pragma unroll
  for (int r = 0; r < 8; ++r) sum[r] = 0.f;
  #pragma unroll
  for (int nt = 0; nt < 8; ++nt) {
    const int n = nt * 16 + lrow;
    const float vn = vvec[n];
    const float qn = qproj[b * Hc + n];
    #pragma unroll
    for (int r = 0; r < 8; ++r) sum[r] += vn * tanhf(qn + acc[nt][r]);
  }
  #pragma unroll
  for (int r = 0; r < 8; ++r) {
    float s = sum[r];
    s += __shfl_xor(s, 1, 32);
    s += __shfl_xor(s, 2, 32);
    s += __shfl_xor(s, 4, 32);
    s += __shfl_xor(s, 8, 32);
    if (lrow == 0) ulog[rowBase + r + lhalf * 8] = s;
  }
}

__global__ void emask_kernel(float* __restrict__ ulog, const int* __restrict__ EI,
                             const int* __restrict__ mask, const int* __restrict__ firstb,
                             const float* __restrict__ scale) {
  size_t total = (size_t)Bc * Ec;
  size_t stride = (size_t)gridDim.x * blockDim.x;
  float sc = scale[0];
  for (size_t i = (size_t)blockIdx.x * blockDim.x + threadIdx.x; i < total; i += stride) {
    int b = (int)(i / Ec), e = (int)(i % Ec);
    int u = EI[(size_t)b * 2 * Ec + e];
    int v = EI[(size_t)b * 2 * Ec + Ec + e];
    int iu = mask[(size_t)b * Nc + u];
    int iv = mask[(size_t)b * Nc + v];
    float em = (firstb[b] || (iu != iv)) ? 0.0f : -INFINITY;
    ulog[i] = sc * tanhf(ulog[i]) + em;
  }
}

__global__ void softmax_kernel(const float* __restrict__ ulog, float* __restrict__ probs) {
  __shared__ float sd[256];
  int b = blockIdx.x, tid = threadIdx.x;
  const float* row = ulog + (size_t)b * Ec;
  float m = -INFINITY;
  for (int e = tid; e < Ec; e += 256) m = fmaxf(m, row[e]);
  sd[tid] = m; __syncthreads();
  for (int k = 128; k > 0; k >>= 1) { if (tid < k) sd[tid] = fmaxf(sd[tid], sd[tid + k]); __syncthreads(); }
  m = sd[0]; __syncthreads();
  float s = 0.f;
  for (int e = tid; e < Ec; e += 256) s += __expf(row[e] - m);
  sd[tid] = s; __syncthreads();
  for (int k = 128; k > 0; k >>= 1) { if (tid < k) sd[tid] += sd[tid + k]; __syncthreads(); }
  float inv = 1.0f / sd[0];
  for (int e = tid; e < Ec; e += 256) probs[(size_t)b * Ec + e] = __expf(row[e] - m) * inv;
}

// ---------------------------------------------------------------------------
// host launcher
// ---------------------------------------------------------------------------
extern "C" void kernel_launch(void* const* d_in, const int* in_sizes, int n_in,
                              void* d_out, int out_size, void* d_ws, size_t ws_size,
                              hipStream_t stream) {
  (void)in_sizes; (void)n_in; (void)out_size; (void)ws_size;

  const int*   lastIdx = (const int*)d_in[0];
  const float* X       = (const float*)d_in[1];
  const float* EF      = (const float*)d_in[2];
  const int*   EI      = (const int*)d_in[3];
  const int*   MaskN   = (const int*)d_in[4];
  const float* enW = (const float*)d_in[5];  const float* enB = (const float*)d_in[6];
  const float* eeW = (const float*)d_in[7];  const float* eeB = (const float*)d_in[8];
  const float* W1 = (const float*)d_in[9];   const float* b1 = (const float*)d_in[10];
  const float* W2 = (const float*)d_in[11];  const float* b2 = (const float*)d_in[12];
  const float* W3 = (const float*)d_in[13];  const float* b3 = (const float*)d_in[14];
  const float* G1 = (const float*)d_in[15];  const float* g1 = (const float*)d_in[16];
  const float* G2 = (const float*)d_in[17];  const float* g2 = (const float*)d_in[18];
  const float* G3 = (const float*)d_in[19];  const float* g3 = (const float*)d_in[20];

  LstmP P;
  P.Wxi = (const float*)d_in[21]; P.bxi = (const float*)d_in[22];
  P.Whi = (const float*)d_in[23]; P.bhi = (const float*)d_in[24];
  P.wci = (const float*)d_in[25]; P.bci = (const float*)d_in[26];
  P.Wxf = (const float*)d_in[27]; P.bxf = (const float*)d_in[28];
  P.Whf = (const float*)d_in[29]; P.bhf = (const float*)d_in[30];
  P.wcf = (const float*)d_in[31]; P.bcf = (const float*)d_in[32];
  P.Wxc = (const float*)d_in[33]; P.bxc = (const float*)d_in[34];
  P.Whc = (const float*)d_in[35]; P.bhc = (const float*)d_in[36];
  P.Wxo = (const float*)d_in[37]; P.bxo = (const float*)d_in[38];
  P.Who = (const float*)d_in[39]; P.bho = (const float*)d_in[40];
  P.wco = (const float*)d_in[41]; P.bco = (const float*)d_in[42];
  const float* Wref = (const float*)d_in[43];
  P.Wq = (const float*)d_in[44];
  const float* vvec = (const float*)d_in[45];
  P.h0 = (const float*)d_in[46];
  P.c0 = (const float*)d_in[47];
  // d_in[48] = start_token (unused in forward)
  const float* r1 = (const float*)d_in[49];
  const float* r2 = (const float*)d_in[50];
  const float* r3 = (const float*)d_in[51];
  const float* ascale = (const float*)d_in[52];

  // outputs (concatenated flat, return order: probs, h, c, final_edge)
  float* out   = (float*)d_out;
  float* probs = out;
  float* hOut  = probs + (size_t)Bc * Ec;
  float* cOut  = hOut + (size_t)Bc * Hc;
  float* FE    = cOut + (size_t)Bc * Hc;          // (B*E) x 128

  // workspace carve (256B aligned)
  const size_t M  = (size_t)Bc * Nc;
  const size_t MH = M * Hc;
  char* w = (char*)d_ws;
  auto carve = [&](size_t bytes) -> void* {
    void* p = (void*)w;
    w += (bytes + 255) & ~(size_t)255;
    return p;
  };
  float*  node32 = (float*)carve(MH * 4);
  float*  nb32   = (float*)carve(MH * 4);
  float*  ctxA32 = (float*)carve(MH * 4);
  float*  ctxB32 = (float*)carve(MH * 4);
  half_t* node16 = (half_t*)carve(MH * 2);
  half_t* nb16   = (half_t*)carve(MH * 2);
  half_t* ctxA16 = (half_t*)carve(MH * 2);
  half_t* ctxB16 = (half_t*)carve(MH * 2);
  half_t* w16    = (half_t*)carve((size_t)7 * Hc * Hc * 2);   // W1,W2,W3,G1,G2,G3,Wref
  float*  cnt    = (float*)carve(M * 4);
  float*  meanx  = (float*)carve((size_t)Bc * Hc * 4);
  float*  qproj  = (float*)carve((size_t)Bc * Hc * 4);
  float*  ulog   = (float*)carve((size_t)Bc * Ec * 4);
  int*    firstb = (int*)carve((size_t)Bc * 4);

  const size_t S = (size_t)Hc * Hc;               // 16384

  // 1) weights -> f16
  const float* wsrc[7] = { W1, W2, W3, G1, G2, G3, Wref };
  for (int k = 0; k < 7; ++k)
    f32_to_f16_kernel<<<(int)((S + 255) / 256), 256, 0, stream>>>(wsrc[k], w16 + k * S, (int)S);

  // 2) embeddings
  embed_nodes_kernel<<<(int)M, Hc, 0, stream>>>(X, enW, enB, node32, node16);
  embed_edges_kernel<<<Bc * Ec, Hc, 0, stream>>>(EF, eeW, eeB, FE);
  node_mean_kernel<<<Bc, Hc, 0, stream>>>(node32, meanx);

  // 3) neighbor aggregation
  zero_f32_kernel<<<2048, 256, 0, stream>>>(nb32, MH);
  zero_f32_kernel<<<256, 256, 0, stream>>>(cnt, M);
  scatter_nb_kernel<<<Bc * Ec, Hc, 0, stream>>>(node32, EI, nb32, cnt);
  normalize_nb_kernel<<<(int)M, Hc, 0, stream>>>(nb32, cnt, nb16);

  // 4) three fused dual-GEMM context layers (WMMA + TDM-staged weights)
  const int gemmBlocks = (int)(M / 128);          // 1024
  ctx_layer_kernel<<<gemmBlocks, 256, 0, stream>>>(node16, nb16, w16 + 0 * S, w16 + 3 * S,
                                                   b1, g1, r1, ctxA32, ctxA16);
  ctx_layer_kernel<<<gemmBlocks, 256, 0, stream>>>(ctxA16, nb16, w16 + 1 * S, w16 + 4 * S,
                                                   b2, g2, r2, ctxB32, ctxB16);
  ctx_layer_kernel<<<gemmBlocks, 256, 0, stream>>>(ctxB16, nb16, w16 + 2 * S, w16 + 5 * S,
                                                   b3, g3, r3, ctxA32, ctxA16);

  // 5) final_edge = edge_emb + ctx[u] + ctx[v]
  gather_final_edge_kernel<<<Bc * Ec, Hc, 0, stream>>>(EI, ctxA32, FE);

  // 6) per-batch "first" flag, LSTM x2 + q projection
  mask_sum_kernel<<<Bc, 256, 0, stream>>>(MaskN, firstb);
  lstm_kernel<<<Bc, Hc, 0, stream>>>(P, meanx, FE, lastIdx, hOut, cOut, qproj);

  // 7) fused attention (WMMA + TDM), mask, softmax
  attn_kernel<<<(Bc * Ec) / 128, 256, 0, stream>>>(FE, w16 + 6 * S, qproj, vvec, ulog);
  emask_kernel<<<2048, 256, 0, stream>>>(ulog, EI, MaskN, firstb, ascale);
  softmax_kernel<<<Bc, 256, 0, stream>>>(ulog, probs);
}
